// PatchNet_40999757807798
// MI455X (gfx1250) — matrix operand order repaired and checked by reading, not compiled, not run on hardware
//
#include <hip/hip_runtime.h>
#include <hip/hip_bf16.h>
#include <stdint.h>

#define CDIV(a,b) (((a)+(b)-1)/(b))
#define EPSBN 1e-5f

typedef __attribute__((ext_vector_type(16))) _Float16 v16h;
typedef __attribute__((ext_vector_type(8)))  float    v8f;

// ================= WMMA GEMM (dominant FLOPs: rl_r / rl_t stacks) =================
// OUT[M x N] = act( A[MxK] @ B[KxN] + bias[M] ), f32 row-major, math f16 WMMA w/ f32 acc.
// One 32x32 output tile per wave32 (2x2 WMMA subtiles, 4 wmma per 32-K step).
// Requires M%32==0, N%32==0, K%32==0.
__device__ __forceinline__ v16h ld_afrag(const float* __restrict__ arow, int k0, int hsel)
{
  // A 16x32 f16 layout: lanes0-15 K={0..7,16..23}, lanes16-31 K={8..15,24..31}
  v16h af;
#pragma unroll
  for (int g = 0; g < 2; ++g) {
    const float4 x0 = *reinterpret_cast<const float4*>(arow + k0 + g * 16 + hsel * 8);
    const float4 x1 = *reinterpret_cast<const float4*>(arow + k0 + g * 16 + hsel * 8 + 4);
    af[g * 8 + 0] = (_Float16)x0.x; af[g * 8 + 1] = (_Float16)x0.y;
    af[g * 8 + 2] = (_Float16)x0.z; af[g * 8 + 3] = (_Float16)x0.w;
    af[g * 8 + 4] = (_Float16)x1.x; af[g * 8 + 5] = (_Float16)x1.y;
    af[g * 8 + 6] = (_Float16)x1.z; af[g * 8 + 7] = (_Float16)x1.w;
  }
  return af;
}

__device__ __forceinline__ v16h ld_bfrag(const float* __restrict__ bcol, int k0, int hsel, int N)
{
  // B 32x16 f16 layout: lanes0-15 K={0..7,16..23}, lanes16-31 K={8..15,24..31}
  v16h bf;
#pragma unroll
  for (int v = 0; v < 8; ++v) {
    const int kb = k0 + (hsel << 3) + ((v >> 2) << 3) + (v << 1);
    bf[2 * v]     = (_Float16)bcol[(size_t)kb * N];
    bf[2 * v + 1] = (_Float16)bcol[(size_t)(kb + 1) * N];
  }
  return bf;
}

__global__ void k_gemm_wmma(const float* __restrict__ A, const float* __restrict__ B,
                            const float* __restrict__ bias, float* __restrict__ C,
                            int M, int N, int K, int relu)
{
  const int lane = threadIdx.x & 31;
  const int wid  = blockIdx.x * (blockDim.x >> 5) + (threadIdx.x >> 5);
  const int tilesN = N >> 5;
  const int tilesM = M >> 5;
  if (wid >= tilesM * tilesN) return;          // uniform per-wave exit: EXEC stays all-1s
  const int tm = wid / tilesN, tn = wid % tilesN;
  const int hsel = lane >> 4, l16 = lane & 15;

  const float* __restrict__ arow0 = A + (size_t)(tm * 32 + l16) * K;
  const float* __restrict__ arow1 = arow0 + (size_t)16 * K;
  const float* __restrict__ bcol0 = B + (size_t)(tn * 32 + l16);
  const float* __restrict__ bcol1 = bcol0 + 16;
  v8f acc00 = {}, acc01 = {}, acc10 = {}, acc11 = {};
  for (int k0 = 0; k0 < K; k0 += 32) {
    __builtin_prefetch(arow0 + k0 + 64, 0, 0);  // global_prefetch_b8
    __builtin_prefetch(arow1 + k0 + 64, 0, 0);
    const v16h a0 = ld_afrag(arow0, k0, hsel);
    const v16h a1 = ld_afrag(arow1, k0, hsel);
    const v16h b0 = ld_bfrag(bcol0, k0, hsel, N);
    const v16h b1 = ld_bfrag(bcol1, k0, hsel, N);
    acc00 = __builtin_amdgcn_wmma_f32_16x16x32_f16(false, a0, false, b0, (short)0, acc00, false, false);
    acc01 = __builtin_amdgcn_wmma_f32_16x16x32_f16(false, a0, false, b1, (short)0, acc01, false, false);
    acc10 = __builtin_amdgcn_wmma_f32_16x16x32_f16(false, a1, false, b0, (short)0, acc10, false, false);
    acc11 = __builtin_amdgcn_wmma_f32_16x16x32_f16(false, a1, false, b1, (short)0, acc11, false, false);
  }
  // C/D layout: vgpr r -> M=r (lanes0-15) / M=r+8 (lanes16-31), N=lane%16
#pragma unroll
  for (int r = 0; r < 8; ++r) {
    const int row0 = tm * 32 + r + hsel * 8;
    const int row1 = row0 + 16;
    const int col0 = tn * 32 + l16;
    float v00 = acc00[r] + bias[row0];
    float v01 = acc01[r] + bias[row0];
    float v10 = acc10[r] + bias[row1];
    float v11 = acc11[r] + bias[row1];
    if (relu) {
      v00 = fmaxf(v00, 0.0f); v01 = fmaxf(v01, 0.0f);
      v10 = fmaxf(v10, 0.0f); v11 = fmaxf(v11, 0.0f);
    }
    C[(size_t)row0 * N + col0]      = v00;
    C[(size_t)row0 * N + col0 + 16] = v01;
    C[(size_t)row1 * N + col0]      = v10;
    C[(size_t)row1 * N + col0 + 16] = v11;
  }
}

// ================= pointwise / utility kernels =================

// Build 33 clouds: cloud0 = x (2000 pts), clouds 1..32 = x gathered by patch_idx (1024 pts).
__global__ void k_gather_xyz(const float* __restrict__ x, const int* __restrict__ pidx,
                             float* __restrict__ xyz, int pmax, int n0, int n, int nclouds)
{
  int tid = blockIdx.x * blockDim.x + threadIdx.x;
  if (tid >= nclouds * pmax) return;
  int p = tid % pmax, c = tid / pmax;
  int N = (c == 0) ? n0 : n;
  if (p >= N) return;
  int src = (c == 0) ? p : pidx[(c - 1) * n + p];
  xyz[(size_t)tid * 3 + 0] = x[(size_t)src * 3 + 0];
  xyz[(size_t)tid * 3 + 1] = x[(size_t)src * 3 + 1];
  xyz[(size_t)tid * 3 + 2] = x[(size_t)src * 3 + 2];
}

// out[(c,p,o)] = (prelu? relu:id)( in[(c,p,:)] . W[o,:] + b[o] )
__global__ void k_linear_point(const float* __restrict__ in, const float* __restrict__ W,
                               const float* __restrict__ b, float* __restrict__ out,
                               int nclouds, int pmax, int n0, int n, int cin, int cout, int prelu)
{
  int tid = blockIdx.x * blockDim.x + threadIdx.x;
  if (tid >= nclouds * pmax * cout) return;
  int o = tid % cout;
  int pc = tid / cout;
  int p = pc % pmax, c = pc / pmax;
  int N = (c == 0) ? n0 : n;
  if (p >= N) return;
  const float* xr = in + ((size_t)c * pmax + p) * cin;
  const float* wr = W + (size_t)o * cin;
  float s = b[o];
  for (int k = 0; k < cin; ++k) s += xr[k] * wr[k];
  if (prelu) s = fmaxf(s, 0.0f);
  out[((size_t)c * pmax + p) * cout + o] = s;
}

// per-(cloud,channel) mean/var over valid points; grid = nclouds*C blocks.
__global__ void k_bn_stats(const float* __restrict__ in, float* __restrict__ mean,
                           float* __restrict__ var, int pmax, int n0, int n, int C)
{
  int c = blockIdx.x / C;
  int ch = blockIdx.x % C;
  int N = (c == 0) ? n0 : n;
  const float* base = in + (size_t)c * pmax * C + ch;
  __shared__ float ss[256], sq[256];
  float s = 0.f, q = 0.f;
  for (int p = threadIdx.x; p < N; p += blockDim.x) {
    float v = base[(size_t)p * C];
    s += v; q += v * v;
  }
  ss[threadIdx.x] = s; sq[threadIdx.x] = q;
  __syncthreads();
  for (int st = blockDim.x >> 1; st > 0; st >>= 1) {
    if (threadIdx.x < st) { ss[threadIdx.x] += ss[threadIdx.x + st]; sq[threadIdx.x] += sq[threadIdx.x + st]; }
    __syncthreads();
  }
  if (threadIdx.x == 0) {
    float m = ss[0] / (float)N;
    mean[blockIdx.x] = m;
    var[blockIdx.x]  = sq[0] / (float)N - m * m;
  }
}

// in-place: h = g*(h-m)*rsqrt(v+eps)+be  (+optional relu)
__global__ void k_bn_apply(float* __restrict__ h, const float* __restrict__ g,
                           const float* __restrict__ be, const float* __restrict__ mean,
                           const float* __restrict__ var, int nclouds, int pmax,
                           int n0, int n, int C, int postrelu)
{
  int tid = blockIdx.x * blockDim.x + threadIdx.x;
  if (tid >= nclouds * pmax * C) return;
  int ch = tid % C;
  int pc = tid / C;
  int p = pc % pmax, c = pc / pmax;
  int N = (c == 0) ? n0 : n;
  if (p >= N) return;
  float m = mean[c * C + ch], v = var[c * C + ch];
  float y = g[ch] * (h[tid] - m) * rsqrtf(v + EPSBN) + be[ch];
  if (postrelu) y = fmaxf(y, 0.0f);
  h[tid] = y;
}

// Farthest-point sampling; one block per cloud. idx[0]=0, then argmax-of-min-dist (ties -> lowest idx).
__global__ void k_fps(const float* __restrict__ xyz, float* __restrict__ dist,
                      int* __restrict__ fidx, float* __restrict__ nxyz,
                      int pmax, int n0, int n, int npoint)
{
  int c = blockIdx.x;
  int N = (c == 0) ? n0 : n;
  const float* p = xyz + (size_t)c * pmax * 3;
  float* d = dist + (size_t)c * pmax;
  int* fi = fidx + (size_t)c * npoint;
  float* nx = nxyz + (size_t)c * npoint * 3;
  __shared__ float sval[256];
  __shared__ int   sidx[256];
  __shared__ float cx, cy, cz;
  __shared__ int   sfar;
  int t = threadIdx.x;
  for (int i = t; i < N; i += 256) d[i] = 1e10f;
  if (t == 0) {
    fi[0] = 0; sfar = 0;
    nx[0] = p[0]; nx[1] = p[1]; nx[2] = p[2];
  }
  __syncthreads();
  for (int it = 1; it < npoint; ++it) {
    if (t == 0) { int f = sfar; cx = p[f * 3]; cy = p[f * 3 + 1]; cz = p[f * 3 + 2]; }
    __syncthreads();
    float best = -1.0f; int bi = 0;
    for (int i = t; i < N; i += 256) {
      float dx = p[i * 3] - cx, dy = p[i * 3 + 1] - cy, dz = p[i * 3 + 2] - cz;
      float dd = dx * dx + dy * dy + dz * dz;
      float m = fminf(d[i], dd);
      d[i] = m;
      if (m > best) { best = m; bi = i; }
    }
    sval[t] = best; sidx[t] = bi;
    __syncthreads();
    for (int st = 128; st > 0; st >>= 1) {
      if (t < st) {
        if (sval[t + st] > sval[t] || (sval[t + st] == sval[t] && sidx[t + st] < sidx[t])) {
          sval[t] = sval[t + st]; sidx[t] = sidx[t + st];
        }
      }
      __syncthreads();
    }
    if (t == 0) {
      sfar = sidx[0];
      fi[it] = sfar;
      nx[it * 3 + 0] = p[sfar * 3 + 0];
      nx[it * 3 + 1] = p[sfar * 3 + 1];
      nx[it * 3 + 2] = p[sfar * 3 + 2];
    }
    __syncthreads();
  }
}

// Ball grouping: first K ascending indices with d^2 <= r^2; pad with first (or N-1 if none).
__global__ void k_ball(const float* __restrict__ xyz, const float* __restrict__ nxyz,
                       int* __restrict__ gidx, int nclouds, int pmax, int n0, int n,
                       int npoint, int K, float r2)
{
  int tid = blockIdx.x * blockDim.x + threadIdx.x;
  if (tid >= nclouds * npoint) return;
  int cent = tid % npoint, c = tid / npoint;
  int N = (c == 0) ? n0 : n;
  const float* p = xyz + (size_t)c * pmax * 3;
  const float* q = nxyz + ((size_t)c * npoint + cent) * 3;
  int* g = gidx + ((size_t)c * npoint + cent) * K;
  float qx = q[0], qy = q[1], qz = q[2];
  int cnt = 0;
  for (int j = 0; j < N && cnt < K; ++j) {
    float dx = p[j * 3] - qx, dy = p[j * 3 + 1] - qy, dz = p[j * 3 + 2] - qz;
    if (dx * dx + dy * dy + dz * dz <= r2) g[cnt++] = j;
  }
  int first = (cnt > 0) ? g[0] : (N - 1);
  for (int k = cnt; k < K; ++k) g[k] = first;
}

// Grouped layer-0 (fused gather): feature = [pts[gi], xyz[gi]-center], per-cloud.
__global__ void k_group_l0(const float* __restrict__ pts, int cpts,
                           const float* __restrict__ xyz, const float* __restrict__ nxyz,
                           const int* __restrict__ gidx, const float* __restrict__ W,
                           const float* __restrict__ b, float* __restrict__ out,
                           int npoint, int K, int cout)
{
  int tid = blockIdx.x * blockDim.x + threadIdx.x;
  if (tid >= npoint * K * cout) return;
  int o = tid % cout;
  int pk = tid / cout;
  int k = pk % K, cent = pk / K;
  int gi = gidx[cent * K + k];
  const float* w = W + (size_t)o * (cpts + 3);
  const float* pp = pts + (size_t)gi * cpts;
  float s = b[o];
  for (int ch = 0; ch < cpts; ++ch) s += pp[ch] * w[ch];
  for (int d = 0; d < 3; ++d) s += (xyz[gi * 3 + d] - nxyz[cent * 3 + d]) * w[cpts + d];
  out[(size_t)pk * cout + o] = s;
}

// max over K samples -> strided (totC) output at channel offset (ptr pre-offset).
__global__ void k_group_max(const float* __restrict__ in, float* __restrict__ out,
                            int npoint, int K, int C, int totC)
{
  int tid = blockIdx.x * blockDim.x + threadIdx.x;
  if (tid >= npoint * C) return;
  int ch = tid % C, cent = tid / C;
  float m = -3.4e38f;
  for (int k = 0; k < K; ++k) m = fmaxf(m, in[((size_t)cent * K + k) * C + ch]);
  out[(size_t)cent * totC + ch] = m;
}

__global__ void k_concat_sa3(const float* __restrict__ xyz, const float* __restrict__ pts,
                             float* __restrict__ out, int nclouds, int P, int cpts)
{
  int cin = cpts + 3;
  int tid = blockIdx.x * blockDim.x + threadIdx.x;
  if (tid >= nclouds * P * cin) return;
  int ch = tid % cin;
  int pc = tid / cin;
  out[tid] = (ch < 3) ? xyz[(size_t)pc * 3 + ch] : pts[(size_t)pc * cpts + (ch - 3)];
}

__global__ void k_max_points(const float* __restrict__ in, float* __restrict__ out,
                             int nclouds, int P, int C)
{
  int tid = blockIdx.x * blockDim.x + threadIdx.x;
  if (tid >= nclouds * C) return;
  int ch = tid % C, c = tid / C;
  float m = -3.4e38f;
  for (int p = 0; p < P; ++p) m = fmaxf(m, in[((size_t)c * P + p) * C + ch]);
  out[tid] = m;
}

__global__ void k_feats(const float* __restrict__ pf, float* __restrict__ feats)
{
  int tid = blockIdx.x * blockDim.x + threadIdx.x;
  if (tid >= 32 * 2048) return;
  int ch = tid % 2048, i = tid / 2048;
  feats[tid] = (ch < 1024) ? pf[ch] : pf[(i + 1) * 1024 + (ch - 1024)];
}

// Exact replication of reference's (32,32,4096) -> (4096,1024) flat reinterpretation.
__global__ void k_rx(const float* __restrict__ feats, float* __restrict__ rx)
{
  int tid = blockIdx.x * blockDim.x + threadIdx.x;
  if (tid >= 4096 * 1024) return;
  int b = tid & 1023;
  int a = tid >> 10;
  int i = a >> 7;
  int j = (a >> 2) & 31;
  int ch = ((a & 3) << 10) | b;
  rx[tid] = (ch < 2048) ? feats[j * 2048 + ch] : feats[i * 2048 + (ch - 2048)];
}

__global__ void k_colmax(const float* __restrict__ in, float* __restrict__ out, int cols)
{
  int r = blockIdx.x;
  __shared__ float s[256];
  float m = -3.4e38f;
  for (int c = threadIdx.x; c < cols; c += blockDim.x) m = fmaxf(m, in[(size_t)r * cols + c]);
  s[threadIdx.x] = m;
  __syncthreads();
  for (int st = 128; st > 0; st >>= 1) {
    if (threadIdx.x < st) s[threadIdx.x] = fmaxf(s[threadIdx.x], s[threadIdx.x + st]);
    __syncthreads();
  }
  if (threadIdx.x == 0) out[r] = s[0];
}

// small FC: act 0=none,1=relu,2=sigmoid
__global__ void k_fc(const float* __restrict__ in, const float* __restrict__ W,
                     const float* __restrict__ b, float* __restrict__ out,
                     int cin, int cout, int act)
{
  int o = blockIdx.x * blockDim.x + threadIdx.x;
  if (o >= cout) return;
  const float* w = W + (size_t)o * cin;
  float s = b[o];
  for (int k = 0; k < cin; ++k) s += in[k] * w[k];
  if (act == 1) s = fmaxf(s, 0.0f);
  else if (act == 2) s = 1.0f / (1.0f + expf(-s));
  out[o] = s;
}

__device__ __forceinline__ bool d_nosym(int i)
{ return i == 0 || i == 1 || i == 2 || i == 3 || i == 4 || i == 7; }

__global__ void k_final(const int* __restrict__ objp, const float* __restrict__ rout,
                        const float* __restrict__ tout, const float* __restrict__ ch,
                        float* __restrict__ out)
{
  if (blockIdx.x || threadIdx.x) return;
  int obj = objp[0];
  int st = 0;
  for (int i = 0; i < obj; ++i) st += d_nosym(i) ? 4 : 12;
  int w = d_nosym(obj) ? 4 : 12;
  int p = 0;
  for (int i = 0; i < w; ++i) out[p++] = rout[st + i];
  for (int k = 0; k < 3; ++k) out[p++] = tout[obj * 3 + k];
  for (int k = 0; k < 3; ++k) out[p++] = ch[k];
}

// ================= host orchestration =================

struct Layer { const float *W, *b, *g, *be; };
struct WB    { const float *W, *b; };

static void run_sa_msg(hipStream_t stream, int nclouds,
                       const float* XYZ, int xyz_pmax, int xyz_n0, int xyz_n,
                       const float* PTS, int cpts, int pts_pmax,
                       float* NXYZ, float* OUTPTS, int totC, int npoint,
                       const float* radii, const int* Ks, const int (*dims)[4],
                       const int* offs, Layer L[3][3],
                       float* DIST, int* FIDX, int* GIDX, float* GA, float* GB,
                       float* GMEAN, float* GVAR)
{
  k_fps<<<nclouds, 256, 0, stream>>>(XYZ, DIST, FIDX, NXYZ, xyz_pmax, xyz_n0, xyz_n, npoint);
  for (int br = 0; br < 3; ++br) {
    const int K = Ks[br];
    const float r2 = radii[br] * radii[br];
    k_ball<<<CDIV(nclouds * npoint, 256), 256, 0, stream>>>(XYZ, NXYZ, GIDX, nclouds,
                                                            xyz_pmax, xyz_n0, xyz_n, npoint, K, r2);
    const int P = npoint * K;
    const int c0 = dims[br][1], c1 = dims[br][2], c2 = dims[br][3];
    for (int c = 0; c < nclouds; ++c) {
      const float* xyzc = XYZ + (size_t)c * xyz_pmax * 3;
      const float* ptsc = PTS + (size_t)c * pts_pmax * cpts;
      const float* nx   = NXYZ + (size_t)c * npoint * 3;
      const int*   gi   = GIDX + (size_t)c * npoint * K;
      // layer 0 (fused gather)
      k_group_l0<<<CDIV(P * c0, 256), 256, 0, stream>>>(ptsc, cpts, xyzc, nx, gi,
                                                        L[br][0].W, L[br][0].b, GA, npoint, K, c0);
      k_bn_stats<<<c0, 256, 0, stream>>>(GA, GMEAN, GVAR, P, P, P, c0);
      k_bn_apply<<<CDIV(P * c0, 256), 256, 0, stream>>>(GA, L[br][0].g, L[br][0].be,
                                                        GMEAN, GVAR, 1, P, P, P, c0, 1);
      // layer 1
      k_linear_point<<<CDIV(P * c1, 256), 256, 0, stream>>>(GA, L[br][1].W, L[br][1].b, GB,
                                                            1, P, P, P, c0, c1, 0);
      k_bn_stats<<<c1, 256, 0, stream>>>(GB, GMEAN, GVAR, P, P, P, c1);
      k_bn_apply<<<CDIV(P * c1, 256), 256, 0, stream>>>(GB, L[br][1].g, L[br][1].be,
                                                        GMEAN, GVAR, 1, P, P, P, c1, 1);
      // layer 2
      k_linear_point<<<CDIV(P * c2, 256), 256, 0, stream>>>(GB, L[br][2].W, L[br][2].b, GA,
                                                            1, P, P, P, c1, c2, 0);
      k_bn_stats<<<c2, 256, 0, stream>>>(GA, GMEAN, GVAR, P, P, P, c2);
      k_bn_apply<<<CDIV(P * c2, 256), 256, 0, stream>>>(GA, L[br][2].g, L[br][2].be,
                                                        GMEAN, GVAR, 1, P, P, P, c2, 1);
      // max over K samples
      float* outc = OUTPTS + (size_t)c * npoint * totC + offs[br];
      k_group_max<<<CDIV(npoint * c2, 256), 256, 0, stream>>>(GA, outc, npoint, K, c2, totC);
    }
  }
}

extern "C" void kernel_launch(void* const* d_in, const int* in_sizes, int n_in,
                              void* d_out, int out_size, void* d_ws, size_t ws_size,
                              hipStream_t stream)
{
  (void)in_sizes; (void)n_in; (void)out_size; (void)ws_size;

  // ---- walk inputs (insertion-order DFS of setup_inputs dict) ----
  int ii = 0;
  auto nf = [&]() -> const float* { return (const float*)d_in[ii++]; };
  const float* X    = nf();
  const int* PIDX   = (const int*)d_in[ii++];
  const int* OBJ    = (const int*)d_in[ii++];
  const float *c1W = nf(), *c1b = nf(), *bn1g = nf(), *bn1be = nf();
  const float *c2W = nf(), *c2b = nf(), *bn2g = nf(), *bn2be = nf();
  Layer sa1L[3][3], sa2L[3][3], sa3L[3];
  for (int br = 0; br < 3; ++br)
    for (int l = 0; l < 3; ++l) { sa1L[br][l].W = nf(); sa1L[br][l].b = nf(); sa1L[br][l].g = nf(); sa1L[br][l].be = nf(); }
  for (int br = 0; br < 3; ++br)
    for (int l = 0; l < 3; ++l) { sa2L[br][l].W = nf(); sa2L[br][l].b = nf(); sa2L[br][l].g = nf(); sa2L[br][l].be = nf(); }
  for (int l = 0; l < 3; ++l) { sa3L[l].W = nf(); sa3L[l].b = nf(); sa3L[l].g = nf(); sa3L[l].be = nf(); }
  WB rlr[3], rlt[3], fcr[2], fcc[2], fct[2];
  for (int l = 0; l < 3; ++l) { rlr[l].W = nf(); rlr[l].b = nf(); }
  for (int l = 0; l < 3; ++l) { rlt[l].W = nf(); rlt[l].b = nf(); }
  for (int l = 0; l < 2; ++l) { fcr[l].W = nf(); fcr[l].b = nf(); }
  for (int l = 0; l < 2; ++l) { fcc[l].W = nf(); fcc[l].b = nf(); }
  for (int l = 0; l < 2; ++l) { fct[l].W = nf(); fct[l].b = nf(); }

  // ---- workspace bump allocator ----
  uint8_t* wsb = (uint8_t*)d_ws;
  size_t off = 0;
  auto alloc = [&](size_t nbytes) -> void* {
    void* p = wsb + off;
    off = (off + nbytes + 255) & ~(size_t)255;
    return p;
  };
  const int NC = 33, PMAX = 2000, N0 = 2000, NP = 1024;
  float* XYZ   = (float*)alloc((size_t)NC * PMAX * 3 * 4);
  float* HA    = (float*)alloc((size_t)NC * PMAX * 128 * 4);
  float* HB    = (float*)alloc((size_t)NC * PMAX * 128 * 4);
  float* MEAN  = (float*)alloc((size_t)NC * 1024 * 4);
  float* VAR   = (float*)alloc((size_t)NC * 1024 * 4);
  float* DIST  = (float*)alloc((size_t)NC * PMAX * 4);
  int*   FIDX  = (int*)  alloc((size_t)NC * 128 * 4);
  int*   GIDX  = (int*)  alloc((size_t)NC * 128 * 128 * 4);
  float* L1XYZ = (float*)alloc((size_t)NC * 128 * 3 * 4);
  float* L1PTS = (float*)alloc((size_t)NC * 128 * 320 * 4);
  float* L2XYZ = (float*)alloc((size_t)NC * 64 * 3 * 4);
  float* L2PTS = (float*)alloc((size_t)NC * 64 * 640 * 4);
  float* GA    = (float*)alloc((size_t)128 * 128 * 128 * 4);
  float* GB    = (float*)alloc((size_t)128 * 128 * 128 * 4);
  float* GMEAN = (float*)alloc(1024 * 4);
  float* GVAR  = (float*)alloc(1024 * 4);
  float* S3IN  = (float*)alloc((size_t)NC * 64 * 643 * 4);
  float* S3A   = (float*)alloc((size_t)NC * 64 * 1024 * 4);
  float* S3B   = (float*)alloc((size_t)NC * 64 * 1024 * 4);
  float* PF    = (float*)alloc((size_t)NC * 1024 * 4);
  float* FEATS = (float*)alloc((size_t)32 * 2048 * 4);
  float* RX    = (float*)alloc((size_t)4096 * 1024 * 4);
  float* R1    = (float*)alloc((size_t)2048 * 1024 * 4);
  float* R2    = (float*)alloc((size_t)1024 * 1024 * 4);
  float* R3    = (float*)alloc((size_t)512 * 1024 * 4);
  float* RXF   = (float*)alloc(512 * 4);
  float* TXF   = (float*)alloc(512 * 4);
  float* T1    = (float*)alloc(512 * 4);
  float* T2    = (float*)alloc(256 * 4);
  float* ROUT  = (float*)alloc(48 * 4);
  float* TOUT  = (float*)alloc(24 * 4);
  float* CH3   = (float*)alloc(3 * 4);

  // ---- build 33 clouds ----
  k_gather_xyz<<<CDIV(NC * PMAX, 256), 256, 0, stream>>>(X, PIDX, XYZ, PMAX, N0, NP, NC);

  // ---- conv1: bn(relu(x@W1.T+b1)) ----
  k_linear_point<<<CDIV(NC * PMAX * 64, 256), 256, 0, stream>>>(XYZ, c1W, c1b, HA, NC, PMAX, N0, NP, 3, 64, 1);
  k_bn_stats<<<NC * 64, 256, 0, stream>>>(HA, MEAN, VAR, PMAX, N0, NP, 64);
  k_bn_apply<<<CDIV(NC * PMAX * 64, 256), 256, 0, stream>>>(HA, bn1g, bn1be, MEAN, VAR, NC, PMAX, N0, NP, 64, 0);
  // ---- conv2 ----
  k_linear_point<<<CDIV(NC * PMAX * 128, 256), 256, 0, stream>>>(HA, c2W, c2b, HB, NC, PMAX, N0, NP, 64, 128, 1);
  k_bn_stats<<<NC * 128, 256, 0, stream>>>(HB, MEAN, VAR, PMAX, N0, NP, 128);
  k_bn_apply<<<CDIV(NC * PMAX * 128, 256), 256, 0, stream>>>(HB, bn2g, bn2be, MEAN, VAR, NC, PMAX, N0, NP, 128, 0);

  // ---- SA1 (multi-scale grouping) ----
  {
    const float radii[3] = {0.1f, 0.2f, 0.4f};
    const int   Ks[3]    = {16, 32, 128};
    const int   dims[3][4] = {{131, 64, 64, 64}, {131, 64, 64, 128}, {131, 64, 96, 128}};
    const int   offs[3]  = {0, 64, 192};
    run_sa_msg(stream, NC, XYZ, PMAX, N0, NP, HB, 128, PMAX, L1XYZ, L1PTS, 320, 128,
               radii, Ks, dims, offs, sa1L, DIST, FIDX, GIDX, GA, GB, GMEAN, GVAR);
  }
  // ---- SA2 ----
  {
    const float radii[3] = {0.2f, 0.4f, 0.8f};
    const int   Ks[3]    = {32, 64, 128};
    const int   dims[3][4] = {{323, 64, 64, 128}, {323, 128, 128, 256}, {323, 128, 128, 256}};
    const int   offs[3]  = {0, 128, 384};
    run_sa_msg(stream, NC, L1XYZ, 128, 128, 128, L1PTS, 320, 128, L2XYZ, L2PTS, 640, 64,
               radii, Ks, dims, offs, sa2L, DIST, FIDX, GIDX, GA, GB, GMEAN, GVAR);
  }

  // ---- SA3 (group-all): relu(bn(h@W.T+b)) x3, max over 64 pts ----
  k_concat_sa3<<<CDIV(NC * 64 * 643, 256), 256, 0, stream>>>(L2XYZ, L2PTS, S3IN, NC, 64, 640);
  k_linear_point<<<CDIV(NC * 64 * 256, 256), 256, 0, stream>>>(S3IN, sa3L[0].W, sa3L[0].b, S3A, NC, 64, 64, 64, 643, 256, 0);
  k_bn_stats<<<NC * 256, 256, 0, stream>>>(S3A, MEAN, VAR, 64, 64, 64, 256);
  k_bn_apply<<<CDIV(NC * 64 * 256, 256), 256, 0, stream>>>(S3A, sa3L[0].g, sa3L[0].be, MEAN, VAR, NC, 64, 64, 64, 256, 1);
  k_linear_point<<<CDIV(NC * 64 * 512, 256), 256, 0, stream>>>(S3A, sa3L[1].W, sa3L[1].b, S3B, NC, 64, 64, 64, 256, 512, 0);
  k_bn_stats<<<NC * 512, 256, 0, stream>>>(S3B, MEAN, VAR, 64, 64, 64, 512);
  k_bn_apply<<<CDIV(NC * 64 * 512, 256), 256, 0, stream>>>(S3B, sa3L[1].g, sa3L[1].be, MEAN, VAR, NC, 64, 64, 64, 512, 1);
  k_linear_point<<<CDIV(NC * 64 * 1024, 256), 256, 0, stream>>>(S3B, sa3L[2].W, sa3L[2].b, S3A, NC, 64, 64, 64, 512, 1024, 0);
  k_bn_stats<<<NC * 1024, 256, 0, stream>>>(S3A, MEAN, VAR, 64, 64, 64, 1024);
  k_bn_apply<<<CDIV(NC * 64 * 1024, 256), 256, 0, stream>>>(S3A, sa3L[2].g, sa3L[2].be, MEAN, VAR, NC, 64, 64, 64, 1024, 1);
  k_max_points<<<CDIV(NC * 1024, 256), 256, 0, stream>>>(S3A, PF, NC, 64, 1024);

  // ---- feats / rx ----
  k_feats<<<CDIV(32 * 2048, 256), 256, 0, stream>>>(PF, FEATS);
  k_rx<<<CDIV(4096 * 1024, 256), 256, 0, stream>>>(FEATS, RX);

  // ---- rl_r stack (WMMA GEMMs, 32x32 tile per wave) + col-max ----
  k_gemm_wmma<<<CDIV((2048 / 32) * (1024 / 32), 8), 256, 0, stream>>>(rlr[0].W, RX, rlr[0].b, R1, 2048, 1024, 4096, 1);
  k_gemm_wmma<<<CDIV((1024 / 32) * (1024 / 32), 8), 256, 0, stream>>>(rlr[1].W, R1, rlr[1].b, R2, 1024, 1024, 2048, 1);
  k_gemm_wmma<<<CDIV((512 / 32) * (1024 / 32), 8), 256, 0, stream>>>(rlr[2].W, R2, rlr[2].b, R3, 512, 1024, 1024, 1);
  k_colmax<<<512, 256, 0, stream>>>(R3, RXF, 1024);
  // ---- rl_t stack ----
  k_gemm_wmma<<<CDIV((2048 / 32) * (1024 / 32), 8), 256, 0, stream>>>(rlt[0].W, RX, rlt[0].b, R1, 2048, 1024, 4096, 1);
  k_gemm_wmma<<<CDIV((1024 / 32) * (1024 / 32), 8), 256, 0, stream>>>(rlt[1].W, R1, rlt[1].b, R2, 1024, 1024, 2048, 1);
  k_gemm_wmma<<<CDIV((512 / 32) * (1024 / 32), 8), 256, 0, stream>>>(rlt[2].W, R2, rlt[2].b, R3, 512, 1024, 1024, 1);
  k_colmax<<<512, 256, 0, stream>>>(R3, TXF, 1024);

  // ---- FC heads ----
  k_fc<<<CDIV(512, 256), 256, 0, stream>>>(RXF, fcr[0].W, fcr[0].b, T1, 512, 512, 1);
  k_fc<<<1, 256, 0, stream>>>(T1, fcr[1].W, fcr[1].b, ROUT, 512, 48, 0);
  k_fc<<<1, 256, 0, stream>>>(RXF, fcc[0].W, fcc[0].b, T2, 512, 256, 1);
  k_fc<<<1, 256, 0, stream>>>(T2, fcc[1].W, fcc[1].b, CH3, 256, 3, 2);
  k_fc<<<1, 256, 0, stream>>>(TXF, fct[0].W, fct[0].b, T2, 512, 256, 1);
  k_fc<<<1, 256, 0, stream>>>(T2, fct[1].W, fct[1].b, TOUT, 256, 24, 0);

  // ---- final assemble: (r_out[st:end], t_out[obj], sigmoid choose) ----
  k_final<<<1, 1, 0, stream>>>(OBJ, ROUT, TOUT, CH3, (float*)d_out);
}
